// ImprovedSchNetInteraction_42468636623526
// MI455X (gfx1250) — matrix operand order repaired and checked by reading, not compiled, not run on hardware
//
#include <hip/hip_runtime.h>

typedef _Float16 v16h __attribute__((ext_vector_type(16)));
typedef float    v8f  __attribute__((ext_vector_type(8)));

#define NN 50000
#define NE 1600000

// ---------- WMMA helpers ----------

__device__ __forceinline__ v8f wmma16(v16h a, v16h b, v8f c) {
    // D = A(16x32 f16) * B(32x16 f16) + C(16x16 f32)
    return __builtin_amdgcn_wmma_f32_16x16x32_f16(false, a, false, b, (short)0, c, false, false);
}

// A fragment from a row-major [16][ld] f16 LDS tile, K-tile kt (32 wide).
// 16-bit A layout: lane m=lane&15; lanes<16 hold K = kb..kb+7 and kb+16..kb+23 with kb = (lane>=16)*8.
__device__ __forceinline__ v16h load_a_frag(const _Float16* tile, int ld, int kt, int lane) {
    int m  = lane & 15;
    int kb = ((lane >> 4) << 3) + (kt << 5);
    v16h r;
    uint4* d = reinterpret_cast<uint4*>(&r);
    d[0] = *reinterpret_cast<const uint4*>(tile + m * ld + kb);
    d[1] = *reinterpret_cast<const uint4*>(tile + m * ld + kb + 16);
    return r;
}

// B fragment from pre-swizzled weights: 512 f16 per (kt,nt) tile, lane-major.
__device__ __forceinline__ v16h load_b_frag(const _Float16* w, int lane) {
    v16h r;
    uint4* d = reinterpret_cast<uint4*>(&r);
    const uint4* s = reinterpret_cast<const uint4*>(w + lane * 16);
    d[0] = s[0];
    d[1] = s[1];
    return r;
}

__device__ __forceinline__ float celu1(float x) { return x > 0.f ? x : (__expf(x) - 1.f); }

// One MLP layer: LDS (f16, row-major) -> LDS (f16, row-major), K = NKT*32, N = NNT*16.
template <int NKT, int NNT, bool ACT>
__device__ __forceinline__ void mlp_layer_lds(const _Float16* in, int ldin,
                                              _Float16* out, int ldout,
                                              const _Float16* w, const float* bias, int lane) {
    int n  = lane & 15;
    int mb = (lane >> 4) << 3;
    v16h afr[NKT];
#pragma unroll
    for (int kt = 0; kt < NKT; ++kt) afr[kt] = load_a_frag(in, ldin, kt, lane);
#pragma unroll
    for (int nt = 0; nt < NNT; ++nt) {
        v8f c = {};
#pragma unroll
        for (int kt = 0; kt < NKT; ++kt)
            c = wmma16(afr[kt], load_b_frag(w + (nt * NKT + kt) * 512, lane), c);
        float bb = bias[nt * 16 + n];
#pragma unroll
        for (int r = 0; r < 8; ++r) {
            float v = c[r] + bb;
            if (ACT) v = celu1(v);
            out[(mb + r) * ldout + nt * 16 + n] = (_Float16)v;
        }
    }
}

// ---------- kernels ----------

// Pack one layer's W[K][N] (f32, row-major) into f16 B fragments.
// grid.x = nkt * (N/16); block tile index b = nt*nkt + kt.
__global__ void __launch_bounds__(256) pack_weights_kernel(const float* __restrict__ W,
                                                           int realK, int nkt, int N,
                                                           _Float16* __restrict__ dst) {
    int b  = blockIdx.x;
    int kt = b % nkt;
    int nt = b / nkt;
    for (int e = threadIdx.x; e < 512; e += 256) {
        int lane = e >> 4, h = e & 15;
        int krow = kt * 32 + ((lane >> 4) << 3) + (h < 8 ? h : h + 8);   // lanes 16..31 hold K+8..K+15, K+24..K+31
        int ncol = nt * 16 + (lane & 15);
        float v  = (krow < realK) ? W[krow * N + ncol] : 0.f;
        dst[b * 512 + e] = (_Float16)v;
    }
}

__global__ void __launch_bounds__(256) zero_kernel(float* __restrict__ p, int n) {
    int i = blockIdx.x * 256 + threadIdx.x;
    if (i < n) p[i] = 0.f;
}

// Pre MLP: x (f32, 50000x32) -> x_pre (f16, 50000x32). [32->64->64->64->32]
__global__ void __launch_bounds__(256) pre_mlp_kernel(const float* __restrict__ x,
                                                      const _Float16* __restrict__ wp,
                                                      const float* __restrict__ b0,
                                                      const float* __restrict__ b1,
                                                      const float* __restrict__ b2,
                                                      const float* __restrict__ b3,
                                                      _Float16* __restrict__ xpre) {
    __shared__ __align__(16) _Float16 sbuf[8][2][1024];
    int lane = threadIdx.x & 31, wv = threadIdx.x >> 5;
    int tile = blockIdx.x * 8 + wv;
    if (tile * 16 >= NN) return;
    _Float16* A = sbuf[wv][0];
    _Float16* B = sbuf[wv][1];
    for (int t = lane; t < 512; t += 32) {
        int row = t >> 5, col = t & 31;
        int node = tile * 16 + row;
        if (node >= NN) node = NN - 1;
        A[row * 32 + col] = (_Float16)x[node * 32 + col];
    }
    mlp_layer_lds<1, 4, true>(A, 32, B, 64, wp + 0, b0, lane);
    mlp_layer_lds<2, 4, true>(B, 64, A, 64, wp + 2048, b1, lane);
    mlp_layer_lds<2, 4, true>(A, 64, B, 64, wp + 6144, b2, lane);
    // last layer K=64 -> N=32, linear, write f16 to global
    int n = lane & 15, mb = (lane >> 4) << 3;
    v16h a0 = load_a_frag(B, 64, 0, lane);
    v16h a1 = load_a_frag(B, 64, 1, lane);
#pragma unroll
    for (int nt = 0; nt < 2; ++nt) {
        v8f c = {};
        c = wmma16(a0, load_b_frag(wp + 10240 + (nt * 2 + 0) * 512, lane), c);
        c = wmma16(a1, load_b_frag(wp + 10240 + (nt * 2 + 1) * 512, lane), c);
        float bb = b3[nt * 16 + n];
#pragma unroll
        for (int r = 0; r < 8; ++r) {
            int node = tile * 16 + mb + r;
            if (node < NN) xpre[node * 32 + nt * 16 + n] = (_Float16)(c[r] + bb);
        }
    }
}

// Edge kernel: bessel -> filter MLP [16(pad32)->64->64->32] -> env scale ->
// messages = x_pre[j]*radial -> atomic scatter-add into agg (= d_out).
__global__ void __launch_bounds__(256) edge_kernel(const float* __restrict__ dist,
                                                   const int* __restrict__ idx,
                                                   const _Float16* __restrict__ wf,
                                                   const float* __restrict__ b0,
                                                   const float* __restrict__ b1,
                                                   const float* __restrict__ b2,
                                                   const _Float16* __restrict__ xpre,
                                                   float* __restrict__ agg) {
    __shared__ __align__(16) _Float16 sbuf[8][2][1024];
    __shared__ float senv[8][16];
    __shared__ int si[8][16], sj[8][16];
    int lane = threadIdx.x & 31, wv = threadIdx.x >> 5;
    int tile = blockIdx.x * 8 + wv;
    if (tile * 16 >= NE) return;
    _Float16* A = sbuf[wv][0];
    _Float16* B = sbuf[wv][1];

    if (lane < 16) {
        int e = tile * 16 + lane;
        bool valid = e < NE;
        if (!valid) e = NE - 1;
        float r  = dist[e];
        float dd = r * 0.2f;
        float d2 = dd * dd, d6 = d2 * d2 * d2;
        float env = 1.f - 28.f * d6 + 48.f * d6 * dd - 21.f * d6 * d2;
        senv[wv][lane] = (valid && r < 5.f) ? env : 0.f;
        si[wv][lane] = idx[e];
        sj[wv][lane] = idx[NE + e];
    }
    {   // bessel features: lane covers edge (lane&15), n = nb+1..nb+8; pad K 16..31 with zeros
        int el = lane & 15;
        int e  = tile * 16 + el;
        if (e >= NE) e = NE - 1;
        float r    = dist[e];
        float invr = __builtin_amdgcn_rcpf(r);
        int nb     = (lane >> 4) << 3;
#pragma unroll
        for (int t = 0; t < 8; ++t) {
            float nv = (float)(nb + t + 1);
            float s  = __sinf(nv * 0.62831853071f * r);
            A[el * 32 + nb + t]      = (_Float16)(0.63245553203f * s * invr);
            A[el * 32 + 16 + nb + t] = (_Float16)0.f;
        }
    }
    mlp_layer_lds<1, 4, true>(A, 32, B, 64, wf + 0, b0, lane);
    mlp_layer_lds<2, 4, true>(B, 64, A, 64, wf + 2048, b1, lane);

    // B is consumed now: gather the 16 x_pre[j] rows (64B each) into B with
    // coalesced b128 loads: 2 lanes per edge row, 32B per lane.
    {
        int el = lane >> 1, half = lane & 1;
        int jj = sj[wv][el];
        const uint4* src = reinterpret_cast<const uint4*>(xpre + jj * 32 + half * 16);
        uint4* dst = reinterpret_cast<uint4*>(B + el * 32 + half * 16);
        dst[0] = src[0];   // 8 halves
        dst[1] = src[1];   // 8 halves (stride 16B within the half-row)
    }

    // last filter layer K=64 -> N=32, keep in registers
    int n = lane & 15, mb = (lane >> 4) << 3;
    v16h a0 = load_a_frag(A, 64, 0, lane);
    v16h a1 = load_a_frag(A, 64, 1, lane);
    v8f c0 = {}, c1 = {};
    c0 = wmma16(a0, load_b_frag(wf + 6144 + 0 * 512, lane), c0);
    c0 = wmma16(a1, load_b_frag(wf + 6144 + 1 * 512, lane), c0);
    c1 = wmma16(a0, load_b_frag(wf + 6144 + 2 * 512, lane), c1);
    c1 = wmma16(a1, load_b_frag(wf + 6144 + 3 * 512, lane), c1);
    float bb0 = b2[n], bb1 = b2[16 + n];
#pragma unroll
    for (int r = 0; r < 8; ++r) {
        int el    = mb + r;
        float env = senv[wv][el];
        int ii = si[wv][el];
        float m0 = (c0[r] + bb0) * env * (float)B[el * 32 + n];
        float m1 = (c1[r] + bb1) * env * (float)B[el * 32 + 16 + n];
        unsafeAtomicAdd(&agg[ii * 32 + n], m0);
        unsafeAtomicAdd(&agg[ii * 32 + 16 + n], m1);
    }
}

// Post MLP, in place on d_out: agg f32 -> out f32. [32->64->64->64->32]
__global__ void __launch_bounds__(256) post_mlp_kernel(float* __restrict__ io,
                                                       const _Float16* __restrict__ wp,
                                                       const float* __restrict__ b0,
                                                       const float* __restrict__ b1,
                                                       const float* __restrict__ b2,
                                                       const float* __restrict__ b3) {
    __shared__ __align__(16) _Float16 sbuf[8][2][1024];
    int lane = threadIdx.x & 31, wv = threadIdx.x >> 5;
    int tile = blockIdx.x * 8 + wv;
    if (tile * 16 >= NN) return;
    _Float16* A = sbuf[wv][0];
    _Float16* B = sbuf[wv][1];
    for (int t = lane; t < 512; t += 32) {
        int row = t >> 5, col = t & 31;
        int node = tile * 16 + row;
        if (node >= NN) node = NN - 1;
        A[row * 32 + col] = (_Float16)io[node * 32 + col];
    }
    mlp_layer_lds<1, 4, true>(A, 32, B, 64, wp + 0, b0, lane);
    mlp_layer_lds<2, 4, true>(B, 64, A, 64, wp + 2048, b1, lane);
    mlp_layer_lds<2, 4, true>(A, 64, B, 64, wp + 6144, b2, lane);
    int n = lane & 15, mb = (lane >> 4) << 3;
    v16h a0 = load_a_frag(B, 64, 0, lane);
    v16h a1 = load_a_frag(B, 64, 1, lane);
#pragma unroll
    for (int nt = 0; nt < 2; ++nt) {
        v8f c = {};
        c = wmma16(a0, load_b_frag(wp + 10240 + (nt * 2 + 0) * 512, lane), c);
        c = wmma16(a1, load_b_frag(wp + 10240 + (nt * 2 + 1) * 512, lane), c);
        float bb = b3[nt * 16 + n];
#pragma unroll
        for (int r = 0; r < 8; ++r) {
            int node = tile * 16 + mb + r;
            if (node < NN) io[node * 32 + nt * 16 + n] = c[r] + bb;
        }
    }
}

// ---------- launch ----------

extern "C" void kernel_launch(void* const* d_in, const int* in_sizes, int n_in,
                              void* d_out, int out_size, void* d_ws, size_t ws_size,
                              hipStream_t stream) {
    (void)in_sizes; (void)n_in; (void)out_size; (void)ws_size;
    const float* x    = (const float*)d_in[0];
    const float* dist = (const float*)d_in[1];
    // filter params: (W,b) x3 ; pre: x4 ; post: x4 ; then neighbour_index
    const float* fW[3] = { (const float*)d_in[2], (const float*)d_in[4], (const float*)d_in[6] };
    const float* fB[3] = { (const float*)d_in[3], (const float*)d_in[5], (const float*)d_in[7] };
    const float* pW[4] = { (const float*)d_in[8], (const float*)d_in[10], (const float*)d_in[12], (const float*)d_in[14] };
    const float* pB[4] = { (const float*)d_in[9], (const float*)d_in[11], (const float*)d_in[13], (const float*)d_in[15] };
    const float* qW[4] = { (const float*)d_in[16], (const float*)d_in[18], (const float*)d_in[20], (const float*)d_in[22] };
    const float* qB[4] = { (const float*)d_in[17], (const float*)d_in[19], (const float*)d_in[21], (const float*)d_in[23] };
    const int* idx = (const int*)d_in[24];

    _Float16* wpack = (_Float16*)d_ws;          // 32768 halves = 64 KB of packed B fragments
    _Float16* xpre  = wpack + 32768;            // 50000*32 f16 = 3.2 MB
    float* out = (float*)d_out;

    // base offsets (in halves): filter @0 (4+8+4 tiles), pre @8192 (4+8+8+4), post @20480
    struct Layer { const float* W; int realK, nkt, N, base; };
    const Layer layers[11] = {
        { fW[0], 16, 1, 64, 0 },     { fW[1], 64, 2, 64, 2048 },  { fW[2], 64, 2, 32, 6144 },
        { pW[0], 32, 1, 64, 8192 },  { pW[1], 64, 2, 64, 10240 }, { pW[2], 64, 2, 64, 14336 },
        { pW[3], 64, 2, 32, 18432 },
        { qW[0], 32, 1, 64, 20480 }, { qW[1], 64, 2, 64, 22528 }, { qW[2], 64, 2, 64, 26624 },
        { qW[3], 64, 2, 32, 30720 },
    };
    for (int l = 0; l < 11; ++l) {
        int ntiles = layers[l].nkt * (layers[l].N / 16);
        pack_weights_kernel<<<ntiles, 256, 0, stream>>>(layers[l].W, layers[l].realK,
                                                        layers[l].nkt, layers[l].N,
                                                        wpack + layers[l].base);
    }

    zero_kernel<<<(NN * 32 + 255) / 256, 256, 0, stream>>>(out, NN * 32);

    pre_mlp_kernel<<<(NN / 16 + 7) / 8, 256, 0, stream>>>(x, wpack + 8192,
                                                          pB[0], pB[1], pB[2], pB[3], xpre);

    edge_kernel<<<(NE / 16 + 7) / 8, 256, 0, stream>>>(dist, idx, wpack,
                                                       fB[0], fB[1], fB[2], xpre, out);

    post_mlp_kernel<<<(NN / 16 + 7) / 8, 256, 0, stream>>>(out, wpack + 20480,
                                                           qB[0], qB[1], qB[2], qB[3]);
}
